// WaveletModel_45672682225694
// MI455X (gfx1250) — matrix operand order repaired and checked by reading, not compiled, not run on hardware
//
#include <hip/hip_runtime.h>
#include <hip/hip_bf16.h>

typedef __attribute__((ext_vector_type(16))) _Float16 v16h;
typedef __attribute__((ext_vector_type(8)))  _Float16 v8h;
typedef __attribute__((ext_vector_type(8)))  float    v8f;
typedef __attribute__((ext_vector_type(4)))  int      v4i;

#define SIG_L     4096
#define NSCALE    16
#define WP_STRIDE 4192          // 16 zeros | 4096 filter | 80 zeros
#define XP_STRIDE 8256          // zero-padded signal row (x lives at [2047,6142])
#define NQ        129           // ceil(4111/32) K-chunks of 32

// ---------------------------------------------------------------------------
// Kernel 1: generate wavelet filters W[s, 0..4095] (faithful to reference),
// L2-normalize per scale, store f16 zero-padded into wp[s*WP_STRIDE + 16 + u].
// ---------------------------------------------------------------------------
__global__ void wavelet_gen(const float* __restrict__ scales,
                            const float* __restrict__ f_mod,
                            const float* __restrict__ poly_mod,
                            const float* __restrict__ exp_mod,
                            const float* __restrict__ frac_order,
                            const float* __restrict__ phase_mod,
                            const int*   __restrict__ ip,
                            _Float16*    __restrict__ wp)
{
    __shared__ float s_rw[SIG_L];
    __shared__ float s_iw[SIG_L];
    __shared__ float red[256];

    const int s   = blockIdx.x;
    const int tid = threadIdx.x;
    const int i   = *ip;

    const float fm = f_mod[i];
    float pm[14];
#pragma unroll
    for (int j = 0; j < 14; ++j) pm[j] = poly_mod[i * 14 + j];
    const float em = exp_mod[i];
    float fo[12];
#pragma unroll
    for (int j = 0; j < 12; ++j) fo[j] = frac_order[i * 12 + j];
    const float ph0 = phase_mod[i * 3 + 0];
    const float ph1 = phase_mod[i * 3 + 1];
    const float ph2 = phase_mod[i * 3 + 2];

    const float PI_F = 3.14159265358979323846f;
    const float f    = 1.0f / scales[s];
    float sigma      = fmaxf(fm / (2.0f * PI_F * f), 1e-6f);
    const float inv2s2 = 1.0f / (2.0f * sigma * sigma);

    float sumsq = 0.0f;
    for (int u = tid; u < SIG_L; u += 256) {
        const float t  = (float)u;
        const float ang = 2.0f * PI_F * f * t;
        const float br = cosf(ang);
        const float bi = sinf(ang);

        float real = br * pm[0];
        float sign = -1.0f;
#pragma unroll
        for (int j = 0; j < 6; ++j) {
            const float c  = (float)(j + 2);
            const float sp = (t > 0.0f) ? powf(t, fo[j]) : 0.0f;
            real += sign * (sp * c / powf(sigma, fo[j]) * c) * br * pm[j + 1];
            sign = -sign;
        }

        float imag = bi * pm[7];
        sign = -1.0f;
#pragma unroll
        for (int j = 6; j < 12; ++j) {
            const float sp = (t > 0.0f) ? powf(t, fo[j]) : 0.0f;
            float term;
            if (j == 9) {
                term = sp * 5.0f / (powf(sigma, fo[j]) * 5.0f) * bi * pm[j + 2];
            } else {
                const float c = (float)(j - 4);
                term = sp * c / powf(sigma, fo[j]) * c * bi * pm[j + 2];
            }
            imag += sign * term;
            sign = -sign;
        }

        const float gauss = em * expf(-t * t * inv2s2);
        const float rw = real * gauss;
        const float iw = imag * gauss;
        s_rw[u] = rw;
        s_iw[u] = iw;
        const float amp = rw + iw;
        sumsq += amp * amp;
    }

    red[tid] = sumsq;
    __syncthreads();
    for (int k = 128; k > 0; k >>= 1) {
        if (tid < k) red[tid] += red[tid + k];
        __syncthreads();
    }
    const float invn = 1.0f / sqrtf(red[0]);

    _Float16* wrow = wp + s * WP_STRIDE;
    for (int u = tid; u < SIG_L; u += 256) {
        const float rw = s_rw[u];
        const float iw = s_iw[u];
        const float samp  = (rw + iw) * invn;
        const float phase = atan2f(iw, rw);
        const float sp = sinf(phase);
        const float cp = cosf(phase);
        const float sc = (fabsf(phase) > 1e-20f) ? (sp / phase) : 1.0f;
        const float w  = (samp * sp * ph0) * (samp * cp * ph1) * (samp * sc * ph2);
        wrow[16 + u] = (_Float16)w;
    }
    if (tid < 16) wrow[tid] = (_Float16)0.0f;
    for (int z = 16 + SIG_L + tid; z < WP_STRIDE; z += 256) wrow[z] = (_Float16)0.0f;
}

// ---------------------------------------------------------------------------
// Kernel 2: x (f32, B*C rows of 4096) -> zero-padded f16 rows, stride 8256,
// signal placed at [2047, 6142]. Removes all bounds checks from the GEMM.
// ---------------------------------------------------------------------------
__global__ void xp_convert(const float* __restrict__ x, _Float16* __restrict__ XP)
{
    const int row = blockIdx.x;                 // row = b*16 + c
    const float* xr = x + (size_t)row * SIG_L;
    _Float16* xo = XP + (size_t)row * XP_STRIDE;
    for (int idx = threadIdx.x; idx < XP_STRIDE; idx += 256) {
        const int m = idx - 2047;
        float v = 0.0f;
        if (m >= 0 && m < SIG_L) v = xr[m];
        xo[idx] = (_Float16)v;
    }
}

// ---------------------------------------------------------------------------
// Kernel 3: materialize Toeplitz B tiles in this GEMM's contiguous-k B-fragment order.
// Lane l: n = l&15, kb = (l>>4)*16; element e -> B[kb+e, n] = wp[16+32q+kb+e-n].
// Layout: BT[((c*NQ + q)*32 + l)*16 + e]  (each lane reads 32 aligned bytes).
// ---------------------------------------------------------------------------
__global__ void btile_gen(const _Float16* __restrict__ wp, _Float16* __restrict__ BT)
{
    const int g = blockIdx.x * 256 + threadIdx.x;   // exactly 16*NQ*32 threads
    const int c   = g / (NQ * 32);
    const int rem = g - c * (NQ * 32);
    const int q   = rem >> 5;
    const int l   = rem & 31;
    const int n   = l & 15;
    const int kb  = (l >> 4) * 16;
    const int z0  = 16 + 32 * q + kb - n;           // always in [1, 4143]
    const _Float16* wrow = wp + c * WP_STRIDE;
    _Float16* o = BT + (size_t)g * 16;
#pragma unroll
    for (int e = 0; e < 16; ++e) o[e] = wrow[z0 + e];
}

// ---------------------------------------------------------------------------
// Kernel 4: correlation as WMMA GEMM.
//   D[b, n] += sum_k A[b,k] * B[k,n],  A = x window (16x32 f16), B = Toeplitz.
// Grid (16 t-blocks, 16 channels), 512 threads = 16 waves, 1 tile per wave.
// All 16 waves share channel c -> stage all NQ B-tiles (132 KB) in LDS once
// via CDNA5 async-to-LDS, then dual-accumulator WMMA loop with LDS B reads.
// ---------------------------------------------------------------------------
#define SBT_CHUNKS (NQ * 32)                 // 4128 chunks of 32 bytes

__global__ void __launch_bounds__(512)
conv_wmma(const _Float16* __restrict__ XP,
          const _Float16* __restrict__ BT,
          float* __restrict__ out)
{
    extern __shared__ _Float16 sBT[];        // NQ * 512 f16 = 132,096 bytes

    const int c    = blockIdx.y;
    const int tb   = blockIdx.x;
    const int wave = threadIdx.x >> 5;
    const int lane = threadIdx.x & 31;
    const int t0   = tb * 256 + wave * 16;

    // ---- stage this channel's Toeplitz tiles into LDS --------------------
    const _Float16* BTc = BT + (size_t)c * (NQ * 512);
    for (int idx = threadIdx.x; idx < SBT_CHUNKS; idx += 512) {
#if __has_builtin(__builtin_amdgcn_global_load_async_to_lds_b128)
        __builtin_amdgcn_global_load_async_to_lds_b128(
            (v4i*)(BTc + (size_t)idx * 16),
            (v4i*)(sBT + (size_t)idx * 16),
            0, 0);
#else
        *(v16h*)(sBT + (size_t)idx * 16) = *(const v16h*)(BTc + (size_t)idx * 16);
#endif
    }
#if __has_builtin(__builtin_amdgcn_global_load_async_to_lds_b128)
#if __has_builtin(__builtin_amdgcn_s_wait_asynccnt)
    __builtin_amdgcn_s_wait_asynccnt(0);
#else
    asm volatile("s_wait_asynccnt 0" ::: "memory");
#endif
#endif
    __syncthreads();

    // ---- WMMA GEMM over the effective K support --------------------------
    const int m  = lane & 15;      // A row (batch b)
    const int kh = lane >> 4;      // K-half select

    const _Float16* arow = XP + (size_t)(m * 16 + c) * XP_STRIDE + t0 + kh * 16;
    const _Float16* sbl  = sBT + (size_t)lane * 16;    // + q*512

    // xp nonzero -> v in [2047-t0, 6142-t0]; W nonzero -> v in [0, 4110]
    int vLo = 2047 - t0; if (vLo < 0)    vLo = 0;
    int vHi = 6142 - t0; if (vHi > 4110) vHi = 4110;
    const int qlo = vLo >> 5;
    const int qhi = vHi >> 5;

    v8f acc0 = {};
    v8f acc1 = {};
    int q = qlo;
    for (; q + 1 <= qhi; q += 2) {
        __builtin_prefetch(arow + (size_t)(q + 4) * 32, 0, 0);
        {
            const _Float16* ap = arow + (size_t)q * 32;
            const v8h alo = *(const v8h*)(ap);
            const v8h ahi = *(const v8h*)(ap + 8);
            const v16h a = __builtin_shufflevector(alo, ahi,
                    0, 1, 2, 3, 4, 5, 6, 7, 8, 9, 10, 11, 12, 13, 14, 15);
            const v16h b = *(const v16h*)(sbl + (size_t)q * 512);
            acc0 = __builtin_amdgcn_wmma_f32_16x16x32_f16(
                    false, a, false, b, (short)0, acc0, false, false);
        }
        {
            const _Float16* ap = arow + (size_t)(q + 1) * 32;
            const v8h alo = *(const v8h*)(ap);
            const v8h ahi = *(const v8h*)(ap + 8);
            const v16h a = __builtin_shufflevector(alo, ahi,
                    0, 1, 2, 3, 4, 5, 6, 7, 8, 9, 10, 11, 12, 13, 14, 15);
            const v16h b = *(const v16h*)(sbl + (size_t)(q + 1) * 512);
            acc1 = __builtin_amdgcn_wmma_f32_16x16x32_f16(
                    false, a, false, b, (short)0, acc1, false, false);
        }
    }
    if (q <= qhi) {
        const _Float16* ap = arow + (size_t)q * 32;
        const v8h alo = *(const v8h*)(ap);
        const v8h ahi = *(const v8h*)(ap + 8);
        const v16h a = __builtin_shufflevector(alo, ahi,
                0, 1, 2, 3, 4, 5, 6, 7, 8, 9, 10, 11, 12, 13, 14, 15);
        const v16h b = *(const v16h*)(sbl + (size_t)q * 512);
        acc0 = __builtin_amdgcn_wmma_f32_16x16x32_f16(
                false, a, false, b, (short)0, acc0, false, false);
    }
    const v8f acc = acc0 + acc1;

    // ---- store: D lanes 0-15 -> N=lane, M=r; lanes 16-31 -> N=lane-16, M=8+r
    const int t  = t0 + (lane & 15);
    const int b0 = 8 * kh;
#pragma unroll
    for (int r = 0; r < 8; ++r) {
        const float v = acc[r];
        float* o = out + ((size_t)((b0 + r) * 256 + c)) * SIG_L + t;
#pragma unroll
        for (int rep = 0; rep < 16; ++rep)      // replicate: out ch j = c + 16*rep
            o[(size_t)rep * (16 * SIG_L)] = v;
    }
}

// ---------------------------------------------------------------------------
extern "C" void kernel_launch(void* const* d_in, const int* in_sizes, int n_in,
                              void* d_out, int out_size, void* d_ws, size_t ws_size,
                              hipStream_t stream)
{
    const float* x      = (const float*)d_in[0];
    const float* scales = (const float*)d_in[1];
    const float* f_mod  = (const float*)d_in[2];
    const float* poly   = (const float*)d_in[3];
    const float* em     = (const float*)d_in[4];
    const float* fo     = (const float*)d_in[5];
    const float* phm    = (const float*)d_in[6];
    const int*   ip     = (const int*)d_in[7];

    char* ws = (char*)d_ws;
    const size_t wp_bytes = (size_t)NSCALE * WP_STRIDE * sizeof(_Float16);   // 134,144
    const size_t xp_bytes = (size_t)256 * XP_STRIDE * sizeof(_Float16);      // 4,227,072
    _Float16* wp = (_Float16*)(ws);
    _Float16* XP = (_Float16*)(ws + wp_bytes);
    _Float16* BT = (_Float16*)(ws + wp_bytes + xp_bytes);                    // 2,113,536

    wavelet_gen<<<NSCALE, 256, 0, stream>>>(scales, f_mod, poly, em, fo, phm, ip, wp);
    xp_convert<<<256, 256, 0, stream>>>(x, XP);
    btile_gen<<<(NSCALE * NQ * 32) / 256, 256, 0, stream>>>(wp, BT);         // 258 blocks
    conv_wmma<<<dim3(16, 16), 512, (NQ * 512) * sizeof(_Float16), stream>>>(
        XP, BT, (float*)d_out);
}
